// ActionDeltaMPredictor_62646392980250
// MI455X (gfx1250) — compile-verified, matrix operand order checked by reading, not run on hardware
//
#include <hip/hip_runtime.h>
#include <hip/hip_bf16.h>
#include <math.h>

// ---------- types ----------
typedef __attribute__((ext_vector_type(16))) __bf16 v16bf;
typedef __attribute__((ext_vector_type(8)))  __bf16 v8bf;
typedef __attribute__((ext_vector_type(4)))  __bf16 v4bf;
typedef __attribute__((ext_vector_type(8)))  float  v8f;
typedef __attribute__((ext_vector_type(4)))  float  v4f;

union ABFrag { v16bf v; v8bf h[2]; };

// ---------- problem constants ----------
constexpr int   H       = 1024;
constexpr int   NROWS   = 131072;
constexpr int   MT      = 32;           // rows per block
constexpr int   ASTR    = H + 8;        // LDS A stride (elements) -> 2064B rows, conflict-free b128
constexpr int   QCOLS   = 256;          // N-quarter width
constexpr int   VALSTR  = QCOLS + 4;    // f32 exchange buffer stride
constexpr int   HQSTR   = QCOLS + 8;    // bf16 h-quarter stride
constexpr float LN_EPS  = 1e-5f;
constexpr float MAX_NORM = 3.0f;

constexpr size_t A_BYTES   = (size_t)MT * ASTR * 2;     // 66048
constexpr size_t VAL_BYTES = (size_t)MT * VALSTR * 4;   // 33280
constexpr size_t HQ_BYTES  = (size_t)MT * HQSTR * 2;    // 16896
constexpr size_t LDS_TOTAL = A_BYTES + VAL_BYTES + HQ_BYTES; // ~116 KB (of 320 KB/WGP)

// ---------- kernel 0: weights f32 -> bf16 ----------
__global__ void cvt_weights_kernel(const float* __restrict__ wg,
                                   const float* __restrict__ wv,
                                   __bf16* __restrict__ wgb,
                                   __bf16* __restrict__ wvb, int n) {
  int i = blockIdx.x * blockDim.x + threadIdx.x;
  if (i < n) {
    wgb[i] = (__bf16)wg[i];
    wvb[i] = (__bf16)wv[i];
  }
}

// ---------- kernel 1: LN + dual WMMA GEMM + SwiGLU + 36-dim projection ----------
__global__ void __launch_bounds__(256)
fused_ln_swiglu_proj_kernel(const float*  __restrict__ x,
                            const float*  __restrict__ gamma,
                            const float*  __restrict__ beta,
                            const __bf16* __restrict__ wgb,
                            const __bf16* __restrict__ wvb,
                            const float*  __restrict__ wout,
                            float*        __restrict__ P) {
  extern __shared__ char smem[];
  __bf16* ldsA = (__bf16*)smem;                              // [MT][ASTR] normalized bf16 rows
  float*  valb = (float*)(smem + A_BYTES);                   // [MT][VALSTR] val-GEMM exchange
  __bf16* hq   = (__bf16*)(smem + A_BYTES + VAL_BYTES);      // [MT][HQSTR] gated h quarter

  const int tid     = threadIdx.x;
  const int wave    = tid >> 5;
  const int lane    = tid & 31;
  const int lane15  = lane & 15;
  const int hiHalf  = lane >> 4;           // 0: lanes 0-15, 1: lanes 16-31
  const int rowBase = blockIdx.x * MT;

  // ---- Stage 0: LayerNorm 32 rows -> LDS (bf16) ----
  for (int r = wave * 4; r < wave * 4 + 4; ++r) {
    const float* xr = x + (size_t)(rowBase + r) * H;
    float s = 0.f, ss = 0.f;
    v4f vals[8];
#pragma unroll
    for (int i = 0; i < 8; ++i) {
      v4f v = *(const v4f*)(xr + i * 128 + lane * 4);
      vals[i] = v;
      s  += v.x + v.y + v.z + v.w;
      ss += v.x * v.x + v.y * v.y + v.z * v.z + v.w * v.w;
    }
#pragma unroll
    for (int m = 16; m >= 1; m >>= 1) {
      s  += __shfl_xor(s,  m, 32);
      ss += __shfl_xor(ss, m, 32);
    }
    const float mu   = s * (1.0f / H);
    const float var  = ss * (1.0f / H) - mu * mu;
    const float rstd = rsqrtf(var + LN_EPS);
#pragma unroll
    for (int i = 0; i < 8; ++i) {
      const int c = i * 128 + lane * 4;
      v4f g = *(const v4f*)(gamma + c);
      v4f b = *(const v4f*)(beta + c);
      v4bf o;
      o.x = (__bf16)((vals[i].x - mu) * rstd * g.x + b.x);
      o.y = (__bf16)((vals[i].y - mu) * rstd * g.y + b.y);
      o.z = (__bf16)((vals[i].z - mu) * rstd * g.z + b.z);
      o.w = (__bf16)((vals[i].w - mu) * rstd * g.w + b.w);
      *(v4bf*)(ldsA + r * ASTR + c) = o;
    }
  }
  __syncthreads();

  // Wave roles: waves 0-3 -> gate GEMM, waves 4-7 -> val GEMM; each owns a 64-col segment.
  const bool   isGate = (wave < 4);
  const int    seg    = wave & 3;
  const __bf16* wptr  = isGate ? wgb : wvb;
  const int    aK     = hiHalf * 8;   // 16-bit A layout: hi lanes hold K+8..15 / K+24..31
  const int    bK     = hiHalf * 8;   // mirrored for B (N across lanes, K contiguous)

  float pAcc[5] = {0.f, 0.f, 0.f, 0.f, 0.f};
  const v8f vzero = {0.f, 0.f, 0.f, 0.f, 0.f, 0.f, 0.f, 0.f};

  for (int q = 0; q < 4; ++q) {
    const int n0 = q * QCOLS + seg * 64;
    v8f acc[2][4];
#pragma unroll
    for (int mt = 0; mt < 2; ++mt)
#pragma unroll
      for (int nt = 0; nt < 4; ++nt) acc[mt][nt] = vzero;

    // ---- Stage 1: K-loop WMMA GEMM (bf16, 16x16x32) ----
    for (int kb = 0; kb < H; kb += 32) {
      ABFrag a0, a1;
      const __bf16* ap0 = ldsA + lane15 * ASTR + kb + aK;
      a0.h[0] = *(const v8bf*)(ap0);
      a0.h[1] = *(const v8bf*)(ap0 + 16);
      const __bf16* ap1 = ap0 + 16 * ASTR;
      a1.h[0] = *(const v8bf*)(ap1);
      a1.h[1] = *(const v8bf*)(ap1 + 16);
#pragma unroll
      for (int nt = 0; nt < 4; ++nt) {
        ABFrag b;
        const __bf16* bp = wptr + (size_t)(n0 + nt * 16 + lane15) * H + kb + bK;
        b.h[0] = *(const v8bf*)(bp);
        b.h[1] = *(const v8bf*)(bp + 16);
        acc[0][nt] = __builtin_amdgcn_wmma_f32_16x16x32_bf16(
            false, a0.v, false, b.v, (short)0, acc[0][nt], false, false);
        acc[1][nt] = __builtin_amdgcn_wmma_f32_16x16x32_bf16(
            false, a1.v, false, b.v, (short)0, acc[1][nt], false, false);
      }
    }

    // ---- Stage 2: val waves publish accumulators to LDS ----
    if (!isGate) {
#pragma unroll
      for (int mt = 0; mt < 2; ++mt)
#pragma unroll
        for (int nt = 0; nt < 4; ++nt)
#pragma unroll
          for (int r = 0; r < 8; ++r) {
            const int row = mt * 16 + r + hiHalf * 8;   // C/D layout: VGPR r -> M=r / M=r+8
            const int col = seg * 64 + nt * 16 + lane15;
            valb[row * VALSTR + col] = acc[mt][nt][r];
          }
    }
    __syncthreads();

    // ---- Stage 3: gate waves apply SwiGLU, write bf16 h quarter ----
    if (isGate) {
#pragma unroll
      for (int mt = 0; mt < 2; ++mt)
#pragma unroll
        for (int nt = 0; nt < 4; ++nt)
#pragma unroll
          for (int r = 0; r < 8; ++r) {
            const int row = mt * 16 + r + hiHalf * 8;
            const int col = seg * 64 + nt * 16 + lane15;
            const float g = acc[mt][nt][r];
            const float v = valb[row * VALSTR + col];
            const float hval = g * v / (1.0f + __expf(-g));   // silu(g)*v
            hq[row * HQSTR + col] = (__bf16)hval;
          }
    }
    __syncthreads();

    // ---- Stage 4: partial projection h_quarter @ w_out^T (32 rows x 36 outs) ----
#pragma unroll
    for (int s = 0; s < 5; ++s) {
      const int idx = tid + s * 256;
      if (idx < MT * 36) {
        const int row = idx / 36;
        const int j   = idx % 36;
        const float*  wj = wout + (size_t)j * H + q * QCOLS;
        const __bf16* hr = hq + row * HQSTR;
        float a = pAcc[s];
#pragma unroll 8
        for (int c = 0; c < QCOLS; ++c) a += (float)hr[c] * wj[c];
        pAcc[s] = a;
      }
    }
    __syncthreads();
  }

  // ---- write projected 6x6 pre-matrices (into d_out; expm runs in place) ----
#pragma unroll
  for (int s = 0; s < 5; ++s) {
    const int idx = tid + s * 256;
    if (idx < MT * 36) {
      const int row = idx / 36;
      const int j   = idx % 36;
      P[(size_t)(rowBase + row) * 36 + j] = pAcc[s];
    }
  }
}

// ---------- kernel 2: skew + frob clip + 6x6 expm, in place ----------
__device__ inline void mm6(const float* A, const float* B, float* C) {
#pragma unroll
  for (int i = 0; i < 6; ++i)
#pragma unroll
    for (int j = 0; j < 6; ++j) {
      float s = 0.f;
#pragma unroll
      for (int k = 0; k < 6; ++k) s += A[i * 6 + k] * B[k * 6 + j];
      C[i * 6 + j] = s;
    }
}

__global__ void __launch_bounds__(256)
expm6_kernel(float* __restrict__ PO, int nrows) {
  const int row = blockIdx.x * blockDim.x + threadIdx.x;
  if (row >= nrows) return;
  float p[36];
#pragma unroll
  for (int i = 0; i < 36; ++i) p[i] = PO[(size_t)row * 36 + i];

  float A[36];
#pragma unroll
  for (int i = 0; i < 6; ++i)
#pragma unroll
    for (int j = 0; j < 6; ++j) A[i * 6 + j] = p[i * 6 + j] - p[j * 6 + i];

  float fr = 0.f;
#pragma unroll
  for (int i = 0; i < 36; ++i) fr += A[i] * A[i];
  fr = sqrtf(fr);
  const float clip = fminf(fr, MAX_NORM) / fmaxf(fr, 1e-8f);

  // scaling & squaring: ||A||_F <= 3 -> B = A/16 (||B|| <= 0.1875), Taylor-8, square 4x
  float Bm[36];
#pragma unroll
  for (int i = 0; i < 36; ++i) Bm[i] = A[i] * (clip * 0.0625f);

  float E[36], T[36], Tn[36];
#pragma unroll
  for (int i = 0; i < 36; ++i) { E[i] = Bm[i]; T[i] = Bm[i]; }
#pragma unroll
  for (int i = 0; i < 6; ++i) E[i * 6 + i] += 1.0f;

#pragma unroll
  for (int k = 2; k <= 8; ++k) {
    mm6(T, Bm, Tn);
    const float invk = 1.0f / (float)k;
#pragma unroll
    for (int i = 0; i < 36; ++i) { T[i] = Tn[i] * invk; E[i] += T[i]; }
  }
#pragma unroll
  for (int sq = 0; sq < 4; ++sq) {
    mm6(E, E, Tn);
#pragma unroll
    for (int i = 0; i < 36; ++i) E[i] = Tn[i];
  }
#pragma unroll
  for (int i = 0; i < 36; ++i) PO[(size_t)row * 36 + i] = E[i];
}

// ---------- host launcher ----------
extern "C" void kernel_launch(void* const* d_in, const int* in_sizes, int n_in,
                              void* d_out, int out_size, void* d_ws, size_t ws_size,
                              hipStream_t stream) {
  const float* x     = (const float*)d_in[0];   // state_token [B,H]
  const float* gamma = (const float*)d_in[1];   // [H]
  const float* beta  = (const float*)d_in[2];   // [H]
  const float* wg    = (const float*)d_in[3];   // w_gate [H,H]
  const float* wv    = (const float*)d_in[4];   // w_val  [H,H]
  const float* wout  = (const float*)d_in[5];   // w_out  [36,H]
  float* out = (float*)d_out;                   // [B,6,6] f32

  __bf16* wgb = (__bf16*)d_ws;
  __bf16* wvb = wgb + (size_t)H * H;

  const int nw = H * H;
  cvt_weights_kernel<<<(nw + 255) / 256, 256, 0, stream>>>(wg, wv, wgb, wvb, nw);

  fused_ln_swiglu_proj_kernel<<<NROWS / MT, 256, LDS_TOTAL, stream>>>(
      x, gamma, beta, wgb, wvb, wout, out);

  expm6_kernel<<<NROWS / 256, 256, 0, stream>>>(out, NROWS);
}
